// PointTransformerUpBlock_5617817224090
// MI455X (gfx1250) — matrix operand order repaired and checked by reading, not compiled, not run on hardware
//
#include <hip/hip_runtime.h>
#include <hip/hip_bf16.h>

// ---------------------------------------------------------------------------
// Problem constants (match reference)
// ---------------------------------------------------------------------------
#define NB    4      // batch
#define NPT   4096   // N high points
#define SPT   1024   // S low points
#define CHIGH 256
#define CLOW  512
#define CMID  64
#define KNB   16     // knn K
#define CAH   256    // attn hidden

typedef __attribute__((ext_vector_type(16))) _Float16 v16h;
typedef __attribute__((ext_vector_type(8)))  _Float16 h8v;
typedef __attribute__((ext_vector_type(4)))  _Float16 h4v;
typedef __attribute__((ext_vector_type(8)))  float    v8f;

__device__ __forceinline__ v8f zero8() {
  v8f z = {0.f, 0.f, 0.f, 0.f, 0.f, 0.f, 0.f, 0.f};
  return z;
}

__device__ __forceinline__ h4v cvt4(float4 t) {
  h4v h;
  h[0] = (_Float16)t.x; h[1] = (_Float16)t.y;
  h[2] = (_Float16)t.z; h[3] = (_Float16)t.w;
  return h;
}

// A fragment (16x32 f16, ISA 7.12.2): lane m = lane&15;
// lanes 0-15 hold K {0..7,16..23}, lanes 16-31 hold K {8..15,24..31}.
__device__ __forceinline__ v16h load_afrag(const _Float16* base, int stride, int lane) {
  const int m  = lane & 15;
  const int kb = (lane >> 4) << 3;           // 0 or 8
  const _Float16* row = base + m * stride;
  h8v lo = *(const h8v*)(row + kb);
  h8v hi = *(const h8v*)(row + 16 + kb);
  return __builtin_shufflevector(lo, hi, 0,1,2,3,4,5,6,7,8,9,10,11,12,13,14,15);
}

// B fragment (32x16 f16): lane col n = lane&15; lanes 0-15 hold K {0..7,16..23},
// lanes 16-31 hold K {8..15,24..31}. base points at [col][k] LDS tile (k contiguous).
__device__ __forceinline__ v16h load_bfrag(const _Float16* base, int stride, int lane) {
  const int n  = lane & 15;
  const int kl = (lane >> 4) << 3;           // 0 or 8
  const _Float16* colp = base + n * stride;
  h8v b0 = *(const h8v*)(colp + kl);
  h8v b1 = *(const h8v*)(colp + 16 + kl);
  return __builtin_shufflevector(b0, b1, 0,1,2,3,4,5,6,7,8,9,10,11,12,13,14,15);
}

__device__ __forceinline__ v8f wmma16(v16h a, v16h b, v8f c) {
  return __builtin_amdgcn_wmma_f32_16x16x32_f16(false, a, false, b, (short)0, c,
                                                false, false);
}

// ---------------------------------------------------------------------------
// Generic 1x1-conv GEMM: Y[M,cols] = W[M,C] @ X[C,cols], batched over grid.z.
// Block tile 64x128, 4 waves (128 thr), K-step 32, double-buffered f16 LDS,
// float4 global loads with batched waits (load phase / store phase split).
// mode 0: +bias; 1: +bias,ReLU; 2: +bias,BN(eval),ReLU; 3: Y += val+bias.
// ---------------------------------------------------------------------------
__global__ __launch_bounds__(128)
void gemm_wmma_kernel(const float* __restrict__ W, const float* __restrict__ X,
                      float* __restrict__ Y, const float* __restrict__ bias,
                      const float* __restrict__ gamma, const float* __restrict__ beta,
                      const float* __restrict__ mean, const float* __restrict__ var,
                      int C, int cols, long strideXb, long strideYb, int mode) {
  __shared__ __align__(16) _Float16 As[2][64 * 40];   // [m][k] stride 40
  __shared__ __align__(16) _Float16 Bs[2][128 * 40];  // [col][k] stride 40
  const int tid  = threadIdx.x;
  const int lane = tid & 31;
  const int wave = tid >> 5;
  const int m0   = blockIdx.y * 64;
  const int col0 = blockIdx.x * 128;
  X += (long)blockIdx.z * strideXb;
  Y += (long)blockIdx.z * strideYb;

  v8f acc[8];
#pragma unroll
  for (int i = 0; i < 8; ++i) acc[i] = zero8();

  // index decomposition is loop-invariant
  int mA[4], kA[4], kB[8], cB[8];
#pragma unroll
  for (int i = 0; i < 4; ++i) { int f = tid + i * 128; mA[i] = f >> 3; kA[i] = (f & 7) * 4; }
#pragma unroll
  for (int i = 0; i < 8; ++i) { int f = tid + i * 128; kB[i] = f >> 5; cB[i] = (f & 31) * 4; }

  float4 ta[4], tb[8];

  auto loadTiles = [&](int k0) {
#pragma unroll
    for (int i = 0; i < 4; ++i)
      ta[i] = *(const float4*)&W[(long)(m0 + mA[i]) * C + (k0 + kA[i])];
#pragma unroll
    for (int i = 0; i < 8; ++i)
      tb[i] = *(const float4*)&X[(long)(k0 + kB[i]) * cols + (col0 + cB[i])];
  };
  auto storeTiles = [&](int buf) {
#pragma unroll
    for (int i = 0; i < 4; ++i)
      *(h4v*)&As[buf][mA[i] * 40 + kA[i]] = cvt4(ta[i]);
#pragma unroll
    for (int i = 0; i < 8; ++i) {
      Bs[buf][(cB[i] + 0) * 40 + kB[i]] = (_Float16)tb[i].x;
      Bs[buf][(cB[i] + 1) * 40 + kB[i]] = (_Float16)tb[i].y;
      Bs[buf][(cB[i] + 2) * 40 + kB[i]] = (_Float16)tb[i].z;
      Bs[buf][(cB[i] + 3) * 40 + kB[i]] = (_Float16)tb[i].w;
    }
  };

  const int nsteps = C >> 5;
  loadTiles(0);
  storeTiles(0);
  __syncthreads();
  for (int s = 0; s < nsteps; ++s) {
    const int cur = s & 1;
    if (s + 1 < nsteps) loadTiles((s + 1) << 5);   // prefetch next k-step
    v16h a = load_afrag(&As[cur][(wave * 16) * 40], 40, lane);
#pragma unroll
    for (int ns = 0; ns < 8; ++ns) {
      v16h bm = load_bfrag(&Bs[cur][(ns * 16) * 40], 40, lane);
      acc[ns] = wmma16(a, bm, acc[ns]);
    }
    if (s + 1 < nsteps) storeTiles(cur ^ 1);
    __syncthreads();
  }

  const int rbase = m0 + wave * 16 + ((lane >> 4) << 3);
#pragma unroll
  for (int ns = 0; ns < 8; ++ns) {
    int col = col0 + ns * 16 + (lane & 15);
#pragma unroll
    for (int r = 0; r < 8; ++r) {
      int row = rbase + r;
      float v = acc[ns][r];
      if (bias) v += bias[row];
      if (mode == 1) {
        v = fmaxf(v, 0.f);
      } else if (mode == 2) {
        float sc = gamma[row] * rsqrtf(var[row] + 1e-5f);
        v = (v - mean[row]) * sc + beta[row];
        v = fmaxf(v, 0.f);
      }
      long o = (long)row * cols + col;
      if (mode == 3) Y[o] += v; else Y[o] = v;
    }
  }
}

// ---------------------------------------------------------------------------
// three_nn: per high point, top-3 nearest low points + normalized 1/d weights
// ---------------------------------------------------------------------------
__global__ __launch_bounds__(256)
void three_nn_kernel(const float* __restrict__ xh, const float* __restrict__ xl,
                     int* __restrict__ idx3, float* __restrict__ w3) {
  int g = blockIdx.x * blockDim.x + threadIdx.x;
  int b = g >> 12, n = g & (NPT - 1);
  const float* xhb = xh + (long)b * 3 * NPT;
  const float* xlb = xl + (long)b * 3 * SPT;
  float qx = xhb[n], qy = xhb[NPT + n], qz = xhb[2 * NPT + n];
  float d0 = 1e30f, d1 = 1e30f, d2 = 1e30f;
  int i0 = 0, i1 = 0, i2 = 0;
  for (int s = 0; s < SPT; ++s) {
    float dx = qx - xlb[s];
    float dy = qy - xlb[SPT + s];
    float dz = qz - xlb[2 * SPT + s];
    float d = dx * dx + dy * dy + dz * dz;
    if (d < d2) {
      if (d < d1) {
        d2 = d1; i2 = i1;
        if (d < d0) { d1 = d0; i1 = i0; d0 = d; i0 = s; }
        else        { d1 = d;  i1 = s; }
      } else { d2 = d; i2 = s; }
    }
  }
  float r0 = 1.f / (d0 + 1e-8f), r1 = 1.f / (d1 + 1e-8f), r2 = 1.f / (d2 + 1e-8f);
  float sm = r0 + r1 + r2;
  idx3[g * 3 + 0] = i0; idx3[g * 3 + 1] = i1; idx3[g * 3 + 2] = i2;
  w3[g * 3 + 0] = r0 / sm; w3[g * 3 + 1] = r1 / sm; w3[g * 3 + 2] = r2 / sm;
}

// ---------------------------------------------------------------------------
// interpolate: points = sum_j w_j * pl[:, idx_j] + points_high  -> d_out
// ---------------------------------------------------------------------------
__global__ __launch_bounds__(256)
void interp_kernel(const float* __restrict__ pl, const int* __restrict__ idx3,
                   const float* __restrict__ w3, const float* __restrict__ ph,
                   float* __restrict__ out) {
  int n = blockIdx.x * 256 + threadIdx.x;
  int c = blockIdx.y, b = blockIdx.z;
  long base = ((long)b * NPT + n) * 3;
  const float* plc = pl + ((long)b * CHIGH + c) * SPT;
  float v = w3[base + 0] * plc[idx3[base + 0]]
          + w3[base + 1] * plc[idx3[base + 1]]
          + w3[base + 2] * plc[idx3[base + 2]];
  long o = ((long)b * CHIGH + c) * NPT + n;
  out[o] = v + ph[o];
}

// ---------------------------------------------------------------------------
// knn: top-16 neighbors among all 4096 high points (register insertion sort).
// ---------------------------------------------------------------------------
__global__ __launch_bounds__(256)
void knn_kernel(const float* __restrict__ xh, int* __restrict__ knn) {
  int g = blockIdx.x * blockDim.x + threadIdx.x;
  int b = g >> 12, n = g & (NPT - 1);
  const float* xb = xh + (long)b * 3 * NPT;
  float qx = xb[n], qy = xb[NPT + n], qz = xb[2 * NPT + n];
  float bd[KNB];
  int bi[KNB];
#pragma unroll
  for (int t = 0; t < KNB; ++t) { bd[t] = 1e30f; bi[t] = 0; }
  for (int j = 0; j < NPT; ++j) {
    float dx = qx - xb[j], dy = qy - xb[NPT + j], dz = qz - xb[2 * NPT + j];
    float d = dx * dx + dy * dy + dz * dz;
    if (d < bd[KNB - 1]) {
      bd[KNB - 1] = d; bi[KNB - 1] = j;
#pragma unroll
      for (int t = KNB - 1; t > 0; --t) {
        if (bd[t] < bd[t - 1]) {
          float td = bd[t]; bd[t] = bd[t - 1]; bd[t - 1] = td;
          int ti = bi[t]; bi[t] = bi[t - 1]; bi[t - 1] = ti;
        }
      }
    }
  }
#pragma unroll
  for (int t = 0; t < KNB; ++t) knn[(long)g * KNB + t] = bi[t];
}

// ---------------------------------------------------------------------------
// Fused PointTransformer attention, one block: per (batch, 8-query tile).
// cols = 8 queries * 16 neighbors = 128 columns of the channel MLPs.
// LDS = 64,000 bytes.
// ---------------------------------------------------------------------------
__global__ __launch_bounds__(256)
void attn_fused_kernel(const float* __restrict__ qkv, const int* __restrict__ knn,
                       const float* __restrict__ xh,
                       const float* __restrict__ pw1, const float* __restrict__ pb1,
                       const float* __restrict__ pw2, const float* __restrict__ pb2,
                       const float* __restrict__ aw1, const float* __restrict__ ab1,
                       const float* __restrict__ aw2, const float* __restrict__ ab2,
                       float* __restrict__ agg) {
  __shared__ __align__(16) _Float16 Us[128 * 72];   // pos-hidden, then u  [col][c]
  __shared__ __align__(16) _Float16 RPs[128 * 72];  // rel_pos             [col][c]
  __shared__ __align__(16) _Float16 S1s[128 * 40];  // s1 32-row chunk     [col][k]
  __shared__ __align__(16) _Float16 AsA[64 * 72];   // A-operand stage (stride 72)
  __shared__ __align__(16) _Float16 AsB[64 * 40];   // A-operand stage (stride 40)
  __shared__ float qs[8 * 64];                      // q[query][c]
  __shared__ int knnS[128];                         // neighbor index per column

  const int tid  = threadIdx.x;
  const int lane = tid & 31;
  const int wave = tid >> 5;                        // 0..7 <-> query in tile
  const int b    = blockIdx.y;
  const int n0   = blockIdx.x * 8;
  const float* qkvb = qkv + (long)b * (3 * CMID) * NPT;
  const float* xhb  = xh + (long)b * 3 * NPT;

  // ---- stage knn indices, q slice, pos_w2 (float4 loads, packed stores) ----
  if (tid < 128) knnS[tid] = knn[((long)b * NPT + n0) * KNB + tid];
#pragma unroll
  for (int i = 0; i < 2; ++i) {
    int e = tid + i * 256;                          // qs: 8*64
    int qloc = e >> 6, c = e & 63;
    qs[e] = qkvb[(long)c * NPT + (n0 + qloc)];
  }
  {
    float4 t[4];
#pragma unroll
    for (int i = 0; i < 4; ++i) t[i] = *(const float4*)&pw2[(tid + i * 256) * 4];
#pragma unroll
    for (int i = 0; i < 4; ++i) {
      int f = tid + i * 256;                        // pos_w2 64x64 = 1024 quads
      int m = f >> 4, k4 = (f & 15) * 4;
      *(h4v*)&AsA[m * 72 + k4] = cvt4(t[i]);
    }
  }
  __syncthreads();

  // ---- pos MLP layer 1 (K=3) -> Us[col][c] ----
  {
    int col = tid & 127;
    int chalf = tid >> 7;
    int q = col >> 4;
    int n = n0 + q;
    int j = knnS[col];
    float rx = xhb[n]           - xhb[j];
    float ry = xhb[NPT + n]     - xhb[NPT + j];
    float rz = xhb[2 * NPT + n] - xhb[2 * NPT + j];
    for (int c = chalf * 32; c < chalf * 32 + 32; ++c) {
      float h = pb1[c] + pw1[c * 3] * rx + pw1[c * 3 + 1] * ry + pw1[c * 3 + 2] * rz;
      Us[col * 72 + c] = (_Float16)fmaxf(h, 0.f);
    }
  }
  __syncthreads();

  // ---- pos conv2 via WMMA: rel_pos = W2 @ h + b2  (M=64,K=64,N=128) ----
#pragma unroll
  for (int tw = 0; tw < 4; ++tw) {
    int mt = tw, nt = wave;
    v8f c = zero8();
#pragma unroll
    for (int ks = 0; ks < 2; ++ks) {
      v16h a  = load_afrag(&AsA[(mt * 16) * 72 + ks * 32], 72, lane);
      v16h bm = load_bfrag(&Us[(nt * 16) * 72 + ks * 32], 72, lane);
      c = wmma16(a, bm, c);
    }
    int col = nt * 16 + (lane & 15);
    int rb = mt * 16 + ((lane >> 4) << 3);
#pragma unroll
    for (int r = 0; r < 8; ++r) {
      int row = rb + r;
      RPs[col * 72 + row] = (_Float16)(c[r] + pb2[row]);
    }
  }
  __syncthreads();

  // ---- u = q - k_gather + rel_pos -> overwrite Us (batched gathers) ----
  for (int i = 0; i < 8; ++i) {
    float g[4];
    int colv[4], cv[4];
#pragma unroll
    for (int jj = 0; jj < 4; ++jj) {
      int e = tid + (i * 4 + jj) * 256;             // 128*64 elements
      int col = e & 127, c = e >> 7;
      colv[jj] = col; cv[jj] = c;
      g[jj] = qkvb[(long)(CMID + c) * NPT + knnS[col]];
    }
#pragma unroll
    for (int jj = 0; jj < 4; ++jj) {
      int col = colv[jj], c = cv[jj];
      int q = col >> 4;
      float u = qs[q * 64 + c] - g[jj] + (float)RPs[col * 72 + c];
      Us[col * 72 + c] = (_Float16)u;
    }
  }
  __syncthreads();

  // ---- attn MLP with 32-row hidden chunking; sim kept in registers ----
  v8f sim[4];
#pragma unroll
  for (int t = 0; t < 4; ++t) sim[t] = zero8();

  for (int hc = 0; hc < 8; ++hc) {
    __syncthreads();
    {
      float4 t1[2], t2[2];
#pragma unroll
      for (int i = 0; i < 2; ++i) {                 // A1 chunk 32x64 = 512 quads
        int f = tid + i * 256;
        int m = f >> 4, k4 = (f & 15) * 4;
        t1[i] = *(const float4*)&aw1[(long)(hc * 32 + m) * CMID + k4];
      }
#pragma unroll
      for (int i = 0; i < 2; ++i) {                 // A2 chunk 64x32 = 512 quads
        int f = tid + i * 256;
        int m = f >> 3, k4 = (f & 7) * 4;
        t2[i] = *(const float4*)&aw2[(long)m * CAH + hc * 32 + k4];
      }
#pragma unroll
      for (int i = 0; i < 2; ++i) {
        int f = tid + i * 256;
        int m = f >> 4, k4 = (f & 15) * 4;
        *(h4v*)&AsA[m * 72 + k4] = cvt4(t1[i]);
      }
#pragma unroll
      for (int i = 0; i < 2; ++i) {
        int f = tid + i * 256;
        int m = f >> 3, k4 = (f & 7) * 4;
        *(h4v*)&AsB[m * 40 + k4] = cvt4(t2[i]);
      }
    }
    __syncthreads();
    // s1 chunk = relu(A1c @ u + b1c): [32][128]
#pragma unroll
    for (int i = 0; i < 2; ++i) {
      int mt2 = i, nt2 = wave;
      v8f c = zero8();
#pragma unroll
      for (int ks = 0; ks < 2; ++ks) {
        v16h a  = load_afrag(&AsA[(mt2 * 16) * 72 + ks * 32], 72, lane);
        v16h bm = load_bfrag(&Us[(nt2 * 16) * 72 + ks * 32], 72, lane);
        c = wmma16(a, bm, c);
      }
      int col = nt2 * 16 + (lane & 15);
      int rb = mt2 * 16 + ((lane >> 4) << 3);
#pragma unroll
      for (int r = 0; r < 8; ++r) {
        int rl = rb + r;
        float v = c[r] + ab1[hc * 32 + rl];
        S1s[col * 40 + rl] = (_Float16)fmaxf(v, 0.f);
      }
    }
    __syncthreads();
    // sim += A2c @ s1c
#pragma unroll
    for (int tw = 0; tw < 4; ++tw) {
      int mt = tw, nt = wave;
      v16h a  = load_afrag(&AsB[(mt * 16) * 40], 40, lane);
      v16h bm = load_bfrag(&S1s[(nt * 16) * 40], 40, lane);
      sim[tw] = wmma16(a, bm, sim[tw]);
    }
  }

  // ---- softmax over K=16 (cross-lane within 16-lane halves) + aggregate ----
  {
    const int colq = lane & 15;                     // = neighbor k
    const int nt = wave;                            // query index in tile
    const int col = nt * 16 + colq;
    const int j = knnS[col];
    const int n = n0 + nt;
#pragma unroll
    for (int tw = 0; tw < 4; ++tw) {
      int rb = tw * 16 + ((lane >> 4) << 3);
      float vv[8];
#pragma unroll
      for (int r = 0; r < 8; ++r) {                 // prefetch v-gathers
        int c = rb + r;
        vv[r] = qkvb[(long)(2 * CMID + c) * NPT + j] + (float)RPs[col * 72 + c];
      }
      float sv[8];
#pragma unroll
      for (int r = 0; r < 8; ++r) sv[r] = sim[tw][r] + ab2[rb + r];
#pragma unroll
      for (int r = 0; r < 8; ++r) {
        float m = sv[r];
        m = fmaxf(m, __shfl_xor(m, 1, 32));
        m = fmaxf(m, __shfl_xor(m, 2, 32));
        m = fmaxf(m, __shfl_xor(m, 4, 32));
        m = fmaxf(m, __shfl_xor(m, 8, 32));
        float e = __expf(sv[r] - m);
        float s = e;
        s += __shfl_xor(s, 1, 32);
        s += __shfl_xor(s, 2, 32);
        s += __shfl_xor(s, 4, 32);
        s += __shfl_xor(s, 8, 32);
        float a = e / s;
        float contrib = a * vv[r];
        contrib += __shfl_xor(contrib, 1, 32);
        contrib += __shfl_xor(contrib, 2, 32);
        contrib += __shfl_xor(contrib, 4, 32);
        contrib += __shfl_xor(contrib, 8, 32);
        if (colq == 0) agg[((long)b * CMID + (rb + r)) * NPT + n] = contrib;
      }
    }
  }
}

// ---------------------------------------------------------------------------
// Host-side orchestration.
// Assumed d_in order = setup_inputs() dict insertion order, params recursed:
//  0 xyz_low  1 xyz_high  2 points_low  3 points_high
//  4..9  up: w,b,gamma,beta,mean,var
//  10..22 block0: before_w,before_b,after_w,after_b,qkv_w,
//                 pos_w1,pos_b1,pos_w2,pos_b2,attn_w1,attn_b1,attn_w2,attn_b2
//  23..35 block1 (same order)
// ---------------------------------------------------------------------------
extern "C" void kernel_launch(void* const* d_in, const int* in_sizes, int n_in,
                              void* d_out, int out_size, void* d_ws, size_t ws_size,
                              hipStream_t stream) {
  const float* xyz_low     = (const float*)d_in[0];
  const float* xyz_high    = (const float*)d_in[1];
  const float* points_low  = (const float*)d_in[2];
  const float* points_high = (const float*)d_in[3];
  const float* up_w    = (const float*)d_in[4];
  const float* up_b    = (const float*)d_in[5];
  const float* up_g    = (const float*)d_in[6];
  const float* up_beta = (const float*)d_in[7];
  const float* up_mean = (const float*)d_in[8];
  const float* up_var  = (const float*)d_in[9];
  float* out = (float*)d_out;

  // workspace layout (floats)
  float* ws    = (float*)d_ws;
  float* pl    = ws;                    // [4,256,1024]  1,048,576
  float* pts   = ws + 1048576;          // [4,64,4096]   1,048,576
  float* qkvb  = ws + 2097152;          // [4,192,4096]  3,145,728
  float* aggb  = ws + 5242880;          // [4,64,4096]   1,048,576
  float* w3    = ws + 6291456;          // [4,4096,3]       49,152
  int*   idx3  = (int*)(ws + 6340608);  // [4,4096,3]       49,152
  int*   knnb  = (int*)(ws + 6389760);  // [4,4096,16]     262,144

  // 1) TransitionUp conv + BN + ReLU (WMMA GEMM, mode 2)
  gemm_wmma_kernel<<<dim3(SPT / 128, CHIGH / 64, NB), 128, 0, stream>>>(
      up_w, points_low, pl, up_b, up_g, up_beta, up_mean, up_var,
      CLOW, SPT, (long)CLOW * SPT, (long)CHIGH * SPT, 2);

  // 2) three_nn weights/indices
  three_nn_kernel<<<(NB * NPT) / 256, 256, 0, stream>>>(xyz_high, xyz_low, idx3, w3);

  // 3) interpolation + skip -> points (d_out)
  interp_kernel<<<dim3(NPT / 256, CHIGH, NB), 256, 0, stream>>>(
      pl, idx3, w3, points_high, out);

  // 4) KNN (shared by both attention blocks)
  knn_kernel<<<(NB * NPT) / 256, 256, 0, stream>>>(xyz_high, knnb);

  for (int blk = 0; blk < 2; ++blk) {
    int bi = 10 + blk * 13;
    const float* before_w = (const float*)d_in[bi + 0];
    const float* before_b = (const float*)d_in[bi + 1];
    const float* after_w  = (const float*)d_in[bi + 2];
    const float* after_b  = (const float*)d_in[bi + 3];
    const float* qkv_w    = (const float*)d_in[bi + 4];
    const float* pos_w1   = (const float*)d_in[bi + 5];
    const float* pos_b1   = (const float*)d_in[bi + 6];
    const float* pos_w2   = (const float*)d_in[bi + 7];
    const float* pos_b2   = (const float*)d_in[bi + 8];
    const float* attn_w1  = (const float*)d_in[bi + 9];
    const float* attn_b1  = (const float*)d_in[bi + 10];
    const float* attn_w2  = (const float*)d_in[bi + 11];
    const float* attn_b2  = (const float*)d_in[bi + 12];

    // before: pts = before_w @ points + before_b
    gemm_wmma_kernel<<<dim3(NPT / 128, CMID / 64, NB), 128, 0, stream>>>(
        before_w, out, pts, before_b, nullptr, nullptr, nullptr, nullptr,
        CHIGH, NPT, (long)CHIGH * NPT, (long)CMID * NPT, 0);

    // qkv = qkv_w @ pts (no bias)
    gemm_wmma_kernel<<<dim3(NPT / 128, (3 * CMID) / 64, NB), 128, 0, stream>>>(
        qkv_w, pts, qkvb, nullptr, nullptr, nullptr, nullptr, nullptr,
        CMID, NPT, (long)CMID * NPT, (long)(3 * CMID) * NPT, 0);

    // fused pos-MLP / attn-MLP / softmax / aggregation
    attn_fused_kernel<<<dim3(NPT / 8, NB), 256, 0, stream>>>(
        qkvb, knnb, xyz_high, pos_w1, pos_b1, pos_w2, pos_b2,
        attn_w1, attn_b1, attn_w2, attn_b2, aggb);

    // after (residual): points += after_w @ agg + after_b
    gemm_wmma_kernel<<<dim3(NPT / 128, CHIGH / 64, NB), 128, 0, stream>>>(
        after_w, aggb, out, after_b, nullptr, nullptr, nullptr, nullptr,
        CMID, NPT, (long)CMID * NPT, (long)CHIGH * NPT, 3);
  }
}
